// PaiNN_72344429134053
// MI455X (gfx1250) — matrix operand order repaired and checked by reading, not compiled, not run on hardware
//
#include <hip/hip_runtime.h>
#include <hip/hip_bf16.h>
#include <math.h>

// ---------------------------------------------------------------------------
// PaiNN forward for MI455X (gfx1250): f16 WMMA for all dense per-atom MLPs,
// VALU + global float atomics for the pair scatter, LDS-chunked WMMA+VALU
// for the rank-3 tensor contraction (sw/dtm/tv) so the 245MB sw tensor is
// never materialized.
//
// v2: LDS operands stored in *fragment order* (each lane's 16 f16 contiguous)
//     so WMMA operand loads are single 32B vector loads (ds_load_b128 x2)
//     instead of 16x ds_load_u16 + v_mov_b16 packing; fast silu/softmax via
//     v_rcp_f32 instead of IEEE divide expansion.
// ---------------------------------------------------------------------------

#define N_RBF   20
#define CUTOFFV 5.0f
#define EPSV    1e-8f

typedef __attribute__((ext_vector_type(16))) _Float16 v16h;
typedef __attribute__((ext_vector_type(8)))  float    v8f;

// ----------------------- WMMA + math helpers -------------------------------
__device__ __forceinline__ v8f wmma_f16(v16h a, v16h b, v8f c) {
  return __builtin_amdgcn_wmma_f32_16x16x32_f16(false, a, false, b, (short)0, c, false, false);
}

__device__ __forceinline__ float fast_rcp(float x) { return __builtin_amdgcn_rcpf(x); }
__device__ __forceinline__ float silu_f(float x) { return x * fast_rcp(1.0f + __expf(-x)); }

// Fragment-order LDS buffers: one 16x32 (A) or 32x16 (B) tile = 512 f16,
// laid out as [lane][16] so each lane's operand is 32 contiguous bytes.
__device__ __forceinline__ v16h load_frag(const _Float16* buf) {
  return *(const v16h*)(buf + (threadIdx.x & 31) * 16);
}

// A-matrix 16x32 f16 element (lane, e) -> k  (CDNA5 ISA 7.12.2 layout)
__device__ __forceinline__ int a_k_of(int lane, int e) {
  return ((lane & 16) >> 1) + ((e & 8) << 1) + (e & 7);
}
// B-matrix 32x16 f16 element (lane, e) -> k
__device__ __forceinline__ int b_k_of(int lane, int e) {
  return ((lane & 16) >> 1) + ((e & 8) << 1) + (e & 7);
}

// Stage 16 rows x (kchunks*32) cols of a row-major f32 matrix into A-fragment
// order ([kc][512]); rows >= nrows padded with zero.
__device__ __forceinline__ void stage_A_frag(_Float16* dst, const float* src,
                                             int row0, int nrows, int ldk,
                                             int kchunks, int tid, int nth) {
  const int total = kchunks * 512;
  for (int i = tid; i < total; i += nth) {
    int kc = i >> 9, lane = (i >> 4) & 31, e = i & 15;
    int m = lane & 15;
    int k = kc * 32 + a_k_of(lane, e);
    int rr = row0 + m;
    dst[i] = (_Float16)((rr < nrows) ? src[(size_t)rr * ldk + k] : 0.0f);
  }
}

// Stage a (kchunks*32) x (ntiles*16) row-major f32 weight matrix into
// B-fragment order ([kc][nt][512]).
__device__ __forceinline__ void stage_B_frag(_Float16* dst, const float* src,
                                             int kchunks, int ntiles, int ldn,
                                             int tid, int nth) {
  const int total = kchunks * ntiles * 512;
  for (int i = tid; i < total; i += nth) {
    int blk = i >> 9;
    int kc = blk / ntiles, nt = blk - kc * ntiles;
    int lane = (i >> 4) & 31, e = i & 15;
    int k = kc * 32 + b_k_of(lane, e);
    int n = nt * 16 + (lane & 15);
    dst[i] = (_Float16)src[(size_t)k * ldn + n];
  }
}

// B-fragment directly from global f32 (for big weights not staged in LDS)
__device__ __forceinline__ v16h frag_b_global_f32(const float* B, int ldn, int nbase) {
  const int lane = threadIdx.x & 31;
  const int n = lane & 15;
  const int kb = (lane < 16) ? 0 : 8;
  v16h b;
#pragma unroll
  for (int v = 0; v < 8; ++v) {
    b[2 * v]     = (_Float16)B[(size_t)(kb + 2 * v + ((2 * v) & 8))     * ldn + nbase + n];
    b[2 * v + 1] = (_Float16)B[(size_t)(kb + 2 * v + 1 + ((2 * v) & 8)) * ldn + nbase + n];
  }
  return b;
}

__device__ __forceinline__ void stage_f32(float* dst, const float* src, int n, int tid, int nth) {
  for (int i = tid; i < n; i += nth) dst[i] = src[i];
}

// ----------------------- init: q = embed[Z], muA = 0 -----------------------
__global__ void embed_init_kernel(const int* __restrict__ Z, const float* __restrict__ embed,
                                  float* __restrict__ q, float* __restrict__ muA, int N) {
  int i = blockIdx.x * blockDim.x + threadIdx.x;
  if (i < N * 64) q[i] = embed[Z[i >> 6] * 64 + (i & 63)];
  if (i < N * 384) muA[i] = 0.0f;
}

// ----------------------- ctx: x = silu(q@W1+b1)@W2+b2 ----------------------
__global__ __launch_bounds__(384) void ctx_kernel(
    const float* __restrict__ q, const float* __restrict__ W1, const float* __restrict__ b1,
    const float* __restrict__ W2, const float* __restrict__ b2, float* __restrict__ x, int N) {
  __shared__ __align__(32) _Float16 qA[2 * 512];        // A frags, K=64
  __shared__ __align__(32) _Float16 W1B[2 * 4 * 512];   // (64,64)
  __shared__ __align__(32) _Float16 W2B[2 * 12 * 512];  // (64,192)
  __shared__ __align__(32) _Float16 hA[2 * 512];        // hidden as A frags
  __shared__ float b1s[64];
  __shared__ float b2s[192];
  const int tid = threadIdx.x, nth = blockDim.x;
  const int a0 = blockIdx.x * 16;
  stage_A_frag(qA, q, a0, N, 64, 2, tid, nth);
  stage_B_frag(W1B, W1, 2, 4, 64, tid, nth);
  stage_B_frag(W2B, W2, 2, 12, 192, tid, nth);
  stage_f32(b1s, b1, 64, tid, nth);
  stage_f32(b2s, b2, 192, tid, nth);
  __syncthreads();
  const int wave = tid >> 5, lane = tid & 31;
  if (wave < 4) {  // hidden: 64 cols, K=64
    v8f acc = {};
    acc = wmma_f16(load_frag(qA),       load_frag(W1B + (0 * 4 + wave) * 512), acc);
    acc = wmma_f16(load_frag(qA + 512), load_frag(W1B + (1 * 4 + wave) * 512), acc);
    const int nn = lane & 15, mb = (lane < 16) ? 0 : 8;
    const int col = wave * 16 + nn;
    const int kc = col >> 5, kk = col & 31;
    const int e = ((kk & 16) >> 1) | (kk & 7);
    const int lb = (kk & 8) ? 16 : 0;
    const float bias = b1s[col];
#pragma unroll
    for (int r = 0; r < 8; ++r)
      hA[kc * 512 + (lb + mb + r) * 16 + e] = (_Float16)silu_f(acc[r] + bias);
  }
  __syncthreads();
  {  // x: 192 cols, K=64, 12 waves
    v8f acc = {};
    acc = wmma_f16(load_frag(hA),       load_frag(W2B + (0 * 12 + wave) * 512), acc);
    acc = wmma_f16(load_frag(hA + 512), load_frag(W2B + (1 * 12 + wave) * 512), acc);
    const int nn = lane & 15, mb = (lane < 16) ? 0 : 8;
    const float bias = b2s[wave * 16 + nn];
#pragma unroll
    for (int r = 0; r < 8; ++r) {
      int atom = a0 + mb + r;
      if (atom < N) x[(size_t)atom * 192 + wave * 16 + nn] = acc[r] + bias;
    }
  }
}

// ----------------------- compress: cv = rowsum(mu) * softmax(MLP(|mu|)) ----
__global__ __launch_bounds__(128) void compress_kernel(
    const float* __restrict__ muA,  // (N,3,128)
    const float* __restrict__ W1, const float* __restrict__ b1,   // (128,32)
    const float* __restrict__ W2, const float* __restrict__ b2,   // (32,64)
    float* __restrict__ cv, int N) {                               // (N,3,64)
  __shared__ __align__(32) _Float16 sA[4 * 512];        // |mu| as A frags, K=128
  __shared__ __align__(32) _Float16 W1B[4 * 2 * 512];   // (128,32)
  __shared__ __align__(32) _Float16 W2B[1 * 4 * 512];   // (32,64)
  __shared__ __align__(32) _Float16 hA[1 * 512];        // hidden A frags, K=32
  __shared__ float b1s[32];
  __shared__ float b2s[64];
  __shared__ float logit[16 * 64];
  __shared__ float rowsum[16 * 3];
  const int tid = threadIdx.x, nth = blockDim.x;
  const int a0 = blockIdx.x * 16;
  stage_B_frag(W1B, W1, 4, 2, 32, tid, nth);
  stage_B_frag(W2B, W2, 1, 4, 64, tid, nth);
  stage_f32(b1s, b1, 32, tid, nth);
  stage_f32(b2s, b2, 64, tid, nth);
  for (int i = tid; i < 4 * 512; i += nth) {  // s = sqrt(sum_i mu^2 + eps)
    int kc = i >> 9, ln = (i >> 4) & 31, e = i & 15;
    int m = ln & 15;
    int c = kc * 32 + a_k_of(ln, e);
    int n = a0 + m;
    float s = 0.0f;
    if (n < N) {
      float v0 = muA[(size_t)(n * 3 + 0) * 128 + c];
      float v1 = muA[(size_t)(n * 3 + 1) * 128 + c];
      float v2 = muA[(size_t)(n * 3 + 2) * 128 + c];
      s = sqrtf(fmaf(v0, v0, fmaf(v1, v1, fmaf(v2, v2, EPSV))));
    }
    sA[i] = (_Float16)s;
  }
  if (tid < 48) {
    int m = tid / 3, i3 = tid % 3, n = a0 + m;
    float s = 0.0f;
    if (n < N)
      for (int c = 0; c < 128; ++c) s += muA[(size_t)(n * 3 + i3) * 128 + c];
    rowsum[m * 3 + i3] = s;
  }
  __syncthreads();
  const int wave = tid >> 5, lane = tid & 31;
  if (wave < 2) {  // hidden 32 cols, K=128
    v8f acc = {};
#pragma unroll
    for (int kc = 0; kc < 4; ++kc)
      acc = wmma_f16(load_frag(sA + kc * 512), load_frag(W1B + (kc * 2 + wave) * 512), acc);
    const int nn = lane & 15, mb = (lane < 16) ? 0 : 8;
    const int col = wave * 16 + nn;       // 0..31 -> single A k-chunk
    const int e = ((col & 16) >> 1) | (col & 7);
    const int lb = (col & 8) ? 16 : 0;
    const float bias = b1s[col];
#pragma unroll
    for (int r = 0; r < 8; ++r)
      hA[(lb + mb + r) * 16 + e] = (_Float16)fmaxf(acc[r] + bias, 0.0f);
  }
  __syncthreads();
  {  // logits 64 cols, K=32, 4 waves
    v8f acc = {};
    acc = wmma_f16(load_frag(hA), load_frag(W2B + wave * 512), acc);
    const int nn = lane & 15, mb = (lane < 16) ? 0 : 8;
    const float bias = b2s[wave * 16 + nn];
#pragma unroll
    for (int r = 0; r < 8; ++r)
      logit[(mb + r) * 64 + wave * 16 + nn] = acc[r] + bias;
  }
  __syncthreads();
  if (tid < 16) {  // softmax per atom row
    float mx = -1e30f;
    for (int f = 0; f < 64; ++f) mx = fmaxf(mx, logit[tid * 64 + f]);
    float sum = 0.0f;
    for (int f = 0; f < 64; ++f) { float e = __expf(logit[tid * 64 + f] - mx); logit[tid * 64 + f] = e; sum += e; }
    float inv = fast_rcp(sum);
    for (int f = 0; f < 64; ++f) logit[tid * 64 + f] *= inv;
  }
  __syncthreads();
  for (int i = tid; i < 16 * 192; i += nth) {
    int m = i / 192, rem = i % 192, i3 = rem >> 6, f = rem & 63, n = a0 + m;
    if (n < N) cv[(size_t)(n * 3 + i3) * 64 + f] = rowsum[m * 3 + i3] * logit[m * 64 + f];
  }
}

// ----------------------- per-pair message + scatter ------------------------
__global__ __launch_bounds__(256) void pair_kernel(
    const float* __restrict__ Rij, const int* __restrict__ idx_i, const int* __restrict__ idx_j,
    const float* __restrict__ fW, const float* __restrict__ fb, int c0,
    const float* __restrict__ x, const float* __restrict__ cv,
    float* __restrict__ dqacc, float* __restrict__ dmuacc, int P) {
  __shared__ float fWs[20 * 192];
  __shared__ float fbs[192];
  const int tid = threadIdx.x, nth = blockDim.x;
  for (int i = tid; i < 20 * 192; i += nth) fWs[i] = fW[(i / 192) * 384 + c0 + (i % 192)];
  for (int i = tid; i < 192; i += nth) fbs[i] = fb[c0 + i];
  __syncthreads();
  const int wave = tid >> 5, lane = tid & 31;
  const int p = blockIdx.x * 8 + wave;
  if (p >= P) return;  // uniform per wave

  float rx = Rij[p * 3 + 0], ry = Rij[p * 3 + 1], rz = Rij[p * 3 + 2];
  float d = sqrtf(rx * rx + ry * ry + rz * rz);
  float inv_d = fast_rcp(d);
  float dirs[3] = {rx * inv_d, ry * inv_d, rz * inv_d};
  const float width = CUTOFFV / (float)(N_RBF - 1);
  float fcut = (d < CUTOFFV) ? 0.5f * (__cosf(3.14159265358979f * d / CUTOFFV) + 1.0f) : 0.0f;
  float phiv = 0.0f;
  if (lane < N_RBF) { float t = d / width - (float)lane; phiv = __expf(-0.5f * t * t); }
  float ph[N_RBF];
#pragma unroll
  for (int k = 0; k < N_RBF; ++k) ph[k] = __shfl(phiv, k, 32);
  const int ai = idx_i[p], aj = idx_j[p];
  float wij[6];
#pragma unroll
  for (int g = 0; g < 6; ++g) {
    int c = lane + 32 * g;
    float acc = fbs[c];
#pragma unroll
    for (int k = 0; k < N_RBF; ++k) acc = fmaf(ph[k], fWs[k * 192 + c], acc);
    wij[g] = acc * fcut;
  }
#pragma unroll
  for (int g = 0; g < 2; ++g) {  // dq scatter
    int f = lane + 32 * g;
    atomicAdd(&dqacc[ai * 64 + f], x[(size_t)aj * 192 + f] * wij[g]);
  }
#pragma unroll
  for (int g = 0; g < 2; ++g) {  // dmu scatter
    int f = lane + 32 * g;
    float xeR = x[(size_t)aj * 192 + 64 + f] * wij[2 + g];
    float xeM = x[(size_t)aj * 192 + 128 + f] * wij[4 + g];
#pragma unroll
    for (int i3 = 0; i3 < 3; ++i3) {
      float val = fmaf(xeR, dirs[i3], xeM * cv[(size_t)(aj * 3 + i3) * 64 + f]);
      atomicAdd(&dmuacc[(size_t)(ai * 3 + i3) * 64 + f], val);
    }
  }
}

// ----------------------- q += dq; mu = dmu + cvi; s_rec --------------------
__global__ void atom_update_kernel(float* __restrict__ q, const float* __restrict__ dqacc,
                                   const float* __restrict__ dmuacc, const float* __restrict__ cv,
                                   float* __restrict__ muN, float* __restrict__ s_rec, int N) {
  int i = blockIdx.x * blockDim.x + threadIdx.x;
  if (i >= N * 64) return;
  int n = i >> 6, f = i & 63;
  q[i] += dqacc[i];
  float ss = EPSV;
#pragma unroll
  for (int i3 = 0; i3 < 3; ++i3) {
    float m = dmuacc[(size_t)(n * 3 + i3) * 64 + f] + cv[(size_t)(n * 3 + i3) * 64 + f];
    muN[(size_t)(n * 3 + i3) * 64 + f] = m;
    ss = fmaf(m, m, ss);
  }
  s_rec[i] = sqrtf(ss);
}

// ----------------------- rec hidden: h = relu(s@W1+b1) ---------------------
__global__ __launch_bounds__(64) void rec_hidden_kernel(
    const float* __restrict__ s_rec, const float* __restrict__ W1, const float* __restrict__ b1,
    float* __restrict__ h, int N) {
  __shared__ __align__(32) _Float16 sA[2 * 512];
  __shared__ __align__(32) _Float16 W1B[2 * 2 * 512];  // (64,32)
  __shared__ float b1s[32];
  const int tid = threadIdx.x, nth = blockDim.x;
  const int a0 = blockIdx.x * 16;
  stage_A_frag(sA, s_rec, a0, N, 64, 2, tid, nth);
  stage_B_frag(W1B, W1, 2, 2, 32, tid, nth);
  stage_f32(b1s, b1, 32, tid, nth);
  __syncthreads();
  const int wave = tid >> 5, lane = tid & 31;
  v8f acc = {};
  acc = wmma_f16(load_frag(sA),       load_frag(W1B + (0 * 2 + wave) * 512), acc);
  acc = wmma_f16(load_frag(sA + 512), load_frag(W1B + (1 * 2 + wave) * 512), acc);
  const int nn = lane & 15, mb = (lane < 16) ? 0 : 8;
  const float bias = b1s[wave * 16 + nn];
#pragma unroll
  for (int r = 0; r < 8; ++r) {
    int a = a0 + mb + r;
    if (a < N) h[(size_t)a * 32 + wave * 16 + nn] = fmaxf(acc[r] + bias, 0.0f);
  }
}

// ----------------------- sw (WMMA, chunked) + Gram contraction -> tv -------
// tv[n,i,r] = sum_c sw[n,r,c] * mu[n,i,c] * (sum_j mu[n,j,c]*mu[n,j,r])
__global__ __launch_bounds__(256) void dtm_tv_kernel(
    const float* __restrict__ muN, const float* __restrict__ h,
    const float* __restrict__ W2, const float* __restrict__ b2,  // (32,4096)
    float* __restrict__ tv, int N) {
  __shared__ float muS[16 * 192];
  __shared__ __align__(32) _Float16 hA[512];   // h as A frags, K=32
  __shared__ float swS[16 * 256];
  const int tid = threadIdx.x, nth = blockDim.x;
  const int a0 = blockIdx.x * 16;
  for (int i = tid; i < 16 * 192; i += nth) {
    int m = i / 192, rem = i % 192, n = a0 + m;
    muS[i] = (n < N) ? muN[(size_t)n * 192 + rem] : 0.0f;
  }
  stage_A_frag(hA, h, a0, N, 32, 1, tid, nth);
  __syncthreads();
  const int wave = tid >> 5, lane = tid & 31;
  for (int chunk = 0; chunk < 16; ++chunk) {
    // sw columns [chunk*256, chunk*256+256): 16 WMMA tiles across 8 waves
#pragma unroll
    for (int t = 0; t < 2; ++t) {
      int tileco = wave * 2 + t;
      int colbase = chunk * 256 + tileco * 16;
      v8f acc = {};
      acc = wmma_f16(load_frag(hA), frag_b_global_f32(W2, 4096, colbase), acc);
      const int nn = lane & 15, mb = (lane < 16) ? 0 : 8;
      const float bias = b2[colbase + nn];
#pragma unroll
      for (int r = 0; r < 8; ++r)
        swS[(mb + r) * 256 + tileco * 16 + nn] = acc[r] + bias;
    }
    __syncthreads();
    if (tid < 64) {  // contract 4 r-values of this chunk, exclusive (m,r) per thread
      int m = tid & 15, rl = tid >> 4;
      int r = chunk * 4 + rl;
      const float* mm = &muS[m * 192];
      float mur0 = mm[r], mur1 = mm[64 + r], mur2 = mm[128 + r];
      float t0 = 0.f, t1 = 0.f, t2 = 0.f;
      for (int c = 0; c < 64; ++c) {
        float sw = swS[m * 256 + rl * 64 + c];
        float m0 = mm[c], m1 = mm[64 + c], m2 = mm[128 + c];
        float g = m0 * mur0 + m1 * mur1 + m2 * mur2;
        float tt = sw * g;
        t0 = fmaf(m0, tt, t0); t1 = fmaf(m1, tt, t1); t2 = fmaf(m2, tt, t2);
      }
      int n = a0 + m;
      if (n < N) {
        tv[(size_t)(n * 3 + 0) * 64 + r] = t0;
        tv[(size_t)(n * 3 + 1) * 64 + r] = t1;
        tv[(size_t)(n * 3 + 2) * 64 + r] = t2;
      }
    }
    __syncthreads();
  }
}

// ----------------------- mu_mix = mu @ Wm (rows = 3N) ----------------------
__global__ __launch_bounds__(256) void mumix_kernel(
    const float* __restrict__ muN, const float* __restrict__ Wm,  // (3N,64),(64,128)
    float* __restrict__ V, int rows) {
  __shared__ __align__(32) _Float16 XA[2 * 512];
  __shared__ __align__(32) _Float16 WmB[2 * 8 * 512];  // (64,128)
  const int tid = threadIdx.x, nth = blockDim.x;
  const int r0 = blockIdx.x * 16;
  stage_A_frag(XA, muN, r0, rows, 64, 2, tid, nth);
  stage_B_frag(WmB, Wm, 2, 8, 128, tid, nth);
  __syncthreads();
  const int wave = tid >> 5, lane = tid & 31;
  v8f acc = {};
  acc = wmma_f16(load_frag(XA),       load_frag(WmB + (0 * 8 + wave) * 512), acc);
  acc = wmma_f16(load_frag(XA + 512), load_frag(WmB + (1 * 8 + wave) * 512), acc);
  const int nn = lane & 15, mb = (lane < 16) ? 0 : 8;
#pragma unroll
  for (int r = 0; r < 8; ++r) {
    int rr = r0 + mb + r;
    if (rr < rows) V[(size_t)rr * 128 + wave * 16 + nn] = acc[r];
  }
}

// ----------------------- mixing MLP: x2 = silu([q,|mu_V|]@W1+b1)@W2+b2 -----
__global__ __launch_bounds__(384) void mix_mlp_kernel(
    const float* __restrict__ q, const float* __restrict__ V,
    const float* __restrict__ W1, const float* __restrict__ b1,  // (128,64)
    const float* __restrict__ W2, const float* __restrict__ b2,  // (64,192)
    float* __restrict__ x2, int N) {
  __shared__ __align__(32) _Float16 aA[4 * 512];        // concat input, K=128
  __shared__ __align__(32) _Float16 W1B[4 * 4 * 512];   // (128,64)
  __shared__ __align__(32) _Float16 W2B[2 * 12 * 512];  // (64,192)
  __shared__ __align__(32) _Float16 hA[2 * 512];
  __shared__ float b1s[64];
  __shared__ float b2s[192];
  const int tid = threadIdx.x, nth = blockDim.x;
  const int a0 = blockIdx.x * 16;
  stage_B_frag(W1B, W1, 4, 4, 64, tid, nth);
  stage_B_frag(W2B, W2, 2, 12, 192, tid, nth);
  stage_f32(b1s, b1, 64, tid, nth);
  stage_f32(b2s, b2, 192, tid, nth);
  for (int i = tid; i < 4 * 512; i += nth) {  // A = concat([q, |mu_V|])
    int kc = i >> 9, ln = (i >> 4) & 31, e = i & 15;
    int m = ln & 15;
    int c = kc * 32 + a_k_of(ln, e);
    int n = a0 + m;
    float val = 0.0f;
    if (n < N) {
      if (c < 64) val = q[(size_t)n * 64 + c];
      else {
        int f = c - 64;
        float v0 = V[(size_t)(n * 3 + 0) * 128 + f];
        float v1 = V[(size_t)(n * 3 + 1) * 128 + f];
        float v2 = V[(size_t)(n * 3 + 2) * 128 + f];
        val = sqrtf(fmaf(v0, v0, fmaf(v1, v1, fmaf(v2, v2, EPSV))));
      }
    }
    aA[i] = (_Float16)val;
  }
  __syncthreads();
  const int wave = tid >> 5, lane = tid & 31;
  if (wave < 4) {  // hidden 64 cols, K=128
    v8f acc = {};
#pragma unroll
    for (int kc = 0; kc < 4; ++kc)
      acc = wmma_f16(load_frag(aA + kc * 512), load_frag(W1B + (kc * 4 + wave) * 512), acc);
    const int nn = lane & 15, mb = (lane < 16) ? 0 : 8;
    const int col = wave * 16 + nn;
    const int kc = col >> 5, kk = col & 31;
    const int e = ((kk & 16) >> 1) | (kk & 7);
    const int lb = (kk & 8) ? 16 : 0;
    const float bias = b1s[col];
#pragma unroll
    for (int r = 0; r < 8; ++r)
      hA[kc * 512 + (lb + mb + r) * 16 + e] = (_Float16)silu_f(acc[r] + bias);
  }
  __syncthreads();
  {  // x2 192 cols, K=64, 12 waves
    v8f acc = {};
    acc = wmma_f16(load_frag(hA),       load_frag(W2B + (0 * 12 + wave) * 512), acc);
    acc = wmma_f16(load_frag(hA + 512), load_frag(W2B + (1 * 12 + wave) * 512), acc);
    const int nn = lane & 15, mb = (lane < 16) ? 0 : 8;
    const float bias = b2s[wave * 16 + nn];
#pragma unroll
    for (int r = 0; r < 8; ++r) {
      int n = a0 + mb + r;
      if (n < N) x2[(size_t)n * 192 + wave * 16 + nn] = acc[r] + bias;
    }
  }
}

// ----------------------- mixing finalize -----------------------------------
__global__ void mix_finalize_kernel(
    float* __restrict__ q, const float* __restrict__ x2, const float* __restrict__ V,
    const float* __restrict__ muN, const float* __restrict__ tv,
    float* __restrict__ muA, int N) {
  int i = blockIdx.x * blockDim.x + threadIdx.x;
  if (i >= N * 64) return;
  int n = i >> 6, f = i & 63;
  float dq = x2[(size_t)n * 192 + f], dmu = x2[(size_t)n * 192 + 64 + f],
        dqmu = x2[(size_t)n * 192 + 128 + f];
  float sv = 0.0f;
#pragma unroll
  for (int i3 = 0; i3 < 3; ++i3) {
    float mv = V[(size_t)(n * 3 + i3) * 128 + f];
    float mw = V[(size_t)(n * 3 + i3) * 128 + 64 + f];
    sv = fmaf(mv, mw, sv);
  }
  q[i] += dq + dqmu * sv;
#pragma unroll
  for (int i3 = 0; i3 < 3; ++i3) {
    float mw = V[(size_t)(n * 3 + i3) * 128 + 64 + f];
    muA[(size_t)(n * 3 + i3) * 128 + f]      = muN[(size_t)(n * 3 + i3) * 64 + f] + dmu * mw;
    muA[(size_t)(n * 3 + i3) * 128 + 64 + f] = tv[(size_t)(n * 3 + i3) * 64 + f];
  }
}

// ----------------------- output copy ---------------------------------------
__global__ void copy_out_kernel(const float* __restrict__ q, const float* __restrict__ muA,
                                float* __restrict__ out, int N) {
  int i = blockIdx.x * blockDim.x + threadIdx.x;
  int nq = N * 64;
  if (i < nq) out[i] = q[i];
  if (i < N * 384) out[nq + i] = muA[i];
}

// ---------------------------------------------------------------------------
extern "C" void kernel_launch(void* const* d_in, const int* in_sizes, int n_in,
                              void* d_out, int out_size, void* d_ws, size_t ws_size,
                              hipStream_t stream) {
  const int*   Z        = (const int*)d_in[0];
  const float* Rij      = (const float*)d_in[1];
  const int*   idx_i    = (const int*)d_in[2];
  const int*   idx_j    = (const int*)d_in[3];
  const float* embed    = (const float*)d_in[5];
  const float* filter_W = (const float*)d_in[6];
  const float* filter_b = (const float*)d_in[7];
  const float* ctx_W1   = (const float*)d_in[8];
  const float* ctx_b1   = (const float*)d_in[9];
  const float* ctx_W2   = (const float*)d_in[10];
  const float* ctx_b2   = (const float*)d_in[11];
  const float* comp_W1  = (const float*)d_in[12];
  const float* comp_b1  = (const float*)d_in[13];
  const float* comp_W2  = (const float*)d_in[14];
  const float* comp_b2  = (const float*)d_in[15];
  const float* rec_W1   = (const float*)d_in[16];
  const float* rec_b1   = (const float*)d_in[17];
  const float* rec_W2   = (const float*)d_in[18];
  const float* rec_b2   = (const float*)d_in[19];
  const float* mix_W1   = (const float*)d_in[20];
  const float* mix_b1   = (const float*)d_in[21];
  const float* mix_W2   = (const float*)d_in[22];
  const float* mix_b2   = (const float*)d_in[23];
  const float* mu_mix_W = (const float*)d_in[24];

  const int N = in_sizes[0];
  const int P = in_sizes[2];

  float* w = (float*)d_ws;
  size_t off = 0;
  auto alloc = [&](size_t n) { float* p = w + off; off += n; return p; };
  float* q     = alloc((size_t)N * 64);
  float* muA   = alloc((size_t)N * 384);
  float* x     = alloc((size_t)N * 192);
  float* cv    = alloc((size_t)N * 192);
  float* dqacc = alloc((size_t)N * 64);
  float* dmu   = alloc((size_t)N * 192);
  float* muN   = alloc((size_t)N * 192);
  float* srec  = alloc((size_t)N * 64);
  float* hrec  = alloc((size_t)N * 32);
  float* tv    = alloc((size_t)N * 192);
  float* V     = alloc((size_t)N * 384);
  float* x2    = alloc((size_t)N * 192);
  (void)ws_size; (void)n_in; (void)out_size;

  const int nb16 = (N + 15) / 16;
  embed_init_kernel<<<(N * 384 + 255) / 256, 256, 0, stream>>>(Z, embed, q, muA, N);

  for (int l = 0; l < 2; ++l) {
    ctx_kernel<<<nb16, 384, 0, stream>>>(q, ctx_W1 + l * 64 * 64, ctx_b1 + l * 64,
                                         ctx_W2 + l * 64 * 192, ctx_b2 + l * 192, x, N);
    compress_kernel<<<nb16, 128, 0, stream>>>(muA, comp_W1 + l * 128 * 32, comp_b1 + l * 32,
                                              comp_W2 + l * 32 * 64, comp_b2 + l * 64, cv, N);
    hipMemsetAsync(dqacc, 0, (size_t)N * 64 * sizeof(float), stream);
    hipMemsetAsync(dmu, 0, (size_t)N * 192 * sizeof(float), stream);
    pair_kernel<<<(P + 7) / 8, 256, 0, stream>>>(Rij, idx_i, idx_j, filter_W, filter_b,
                                                 l * 192, x, cv, dqacc, dmu, P);
    atom_update_kernel<<<(N * 64 + 255) / 256, 256, 0, stream>>>(q, dqacc, dmu, cv, muN, srec, N);
    rec_hidden_kernel<<<nb16, 64, 0, stream>>>(srec, rec_W1 + l * 64 * 32, rec_b1 + l * 32, hrec, N);
    dtm_tv_kernel<<<nb16, 256, 0, stream>>>(muN, hrec, rec_W2 + (size_t)l * 32 * 4096,
                                            rec_b2 + l * 4096, tv, N);
    mumix_kernel<<<(3 * N + 15) / 16, 256, 0, stream>>>(muN, mu_mix_W + l * 64 * 128, V, 3 * N);
    mix_mlp_kernel<<<nb16, 384, 0, stream>>>(q, V, mix_W1 + l * 128 * 64, mix_b1 + l * 64,
                                             mix_W2 + l * 64 * 192, mix_b2 + l * 192, x2, N);
    mix_finalize_kernel<<<(N * 64 + 255) / 256, 256, 0, stream>>>(q, x2, V, muN, tv, muA, N);
  }

  copy_out_kernel<<<(N * 384 + 255) / 256, 256, 0, stream>>>(q, muA, (float*)d_out, N);
}
